// RNNTDecodingWrapper_48129403519319
// MI455X (gfx1250) — matrix operand order repaired and checked by reading, not compiled, not run on hardware
//
#include <hip/hip_runtime.h>
#include <hip/hip_bf16.h>

// ---------------------------------------------------------------------------
// RNN-T joint decoding for MI455X (gfx1250), bf16 WMMA everywhere it matters.
// B=8, T=200, U=100 (U1=101), V=1024, E=H=DENC=JH=512, 4H=2048.  K is 512 in
// every GEMM -> ktiles=16 hardcoded so B-fragment addresses constant-fold.
// Per k-step: all B-fragment loads issue first, then the WMMAs (partial waits).
// ---------------------------------------------------------------------------

#define BB    8
#define TT    200
#define U1    101
#define VV    1024
#define HH    512
#define G4H   2048
#define KK512 512
#define KT    16      // K / 32

typedef __attribute__((ext_vector_type(16))) __bf16 v16bf;
typedef __attribute__((ext_vector_type(8)))  float  v8f;

union Frag {
    unsigned int u[8];
    v16bf v;
};

#define WMMA_BF16(A, Bf, C) \
    __builtin_amdgcn_wmma_f32_16x16x32_bf16(false, (A), false, (Bf), (short)0, (C), false, false)

__device__ __forceinline__ unsigned int pack_bf16(float lo, float hi) {
    // CDNA5 keeps the CVT_PK family (cf. V_CVT_PK_FP8_F32 in ISA §7.12.6);
    // clang declares no builtin for bf16 on gfx1250, so emit it directly.
    unsigned int r;
    asm("v_cvt_pk_bf16_f32 %0, %1, %2" : "=v"(r) : "v"(lo), "v"(hi));
    return r;
}

__device__ __forceinline__ float fsigmoid(float x) {
    return __builtin_amdgcn_rcpf(1.f + __expf(-x));
}
__device__ __forceinline__ float ftanh(float x) {      // 1 - 2/(e^{2x}+1)
    float e = __expf(2.f * x);
    return 1.f - 2.f * __builtin_amdgcn_rcpf(e + 1.f);
}

// ---------------------------------------------------------------------------
// Pack a row-major f32 weight (N,K) into WMMA B-fragment tiles:
//   dst[ntile][ktile][lane][r] (dwords); k = 2r + p + 8*(r>>2) + 8*(lane>>4), n = ntile*16 + (lane&15)
// grid = (N/16, K/32), block = 32
// ---------------------------------------------------------------------------
__global__ void pack_b_kernel(const float* __restrict__ W, unsigned int* __restrict__ dst,
                              int N, int K) {
    const int lane  = threadIdx.x;
    const int ntile = blockIdx.x;
    const int ktile = blockIdx.y;
    const int n     = ntile * 16 + (lane & 15);
    const int half  = lane >> 4;
    const float* row = W + (size_t)n * K + ktile * 32 + 8 * half;
    unsigned int* o  = dst + ((size_t)ntile * (K >> 5) + ktile) * 256 + lane * 8;
#pragma unroll
    for (int r = 0; r < 8; ++r)
        o[r] = pack_bf16(row[2 * r + 8 * (r >> 2)], row[2 * r + 8 * (r >> 2) + 1]);
}

// ---------------------------------------------------------------------------
// x = [SOS ; embed[targets]]  -> (B, U1, 512) f32
// ---------------------------------------------------------------------------
__global__ void build_x_kernel(const int* __restrict__ targets, const float* __restrict__ embed,
                               float* __restrict__ x) {
    long i = (long)blockIdx.x * blockDim.x + threadIdx.x;
    if (i >= (long)BB * U1 * HH) return;
    int e    = (int)(i & 511);
    long row = i >> 9;
    int uu   = (int)(row % U1);
    int b    = (int)(row / U1);
    float v  = 0.f;
    if (uu > 0) {
        int t = targets[b * 100 + (uu - 1)];
        v = embed[(size_t)t * 512 + e];
    }
    x[i] = v;
}

// enc (B, DENC, T) -> encT (B*T, DENC)
__global__ void transpose_enc_kernel(const float* __restrict__ enc, float* __restrict__ encT) {
    long i = (long)blockIdx.x * blockDim.x + threadIdx.x;
    if (i >= (long)BB * TT * HH) return;
    int d    = (int)(i & 511);
    long row = i >> 9;
    int t    = (int)(row % TT);
    int b    = (int)(row / TT);
    encT[i] = enc[((size_t)b * 512 + d) * TT + t];
}

// ---------------------------------------------------------------------------
// Generic GEMM (K=512): C(M,N) = A(M,512) @ Wpacked^T + bias, bf16 WMMA, f32 acc.
// block = 128 (4 waves), wave tile = 16x64, WG tile = 16x256.
// grid = (N/256, ceil(M/16))
// ---------------------------------------------------------------------------
__global__ void gemm_bf16_kernel(const float* __restrict__ A, const unsigned int* __restrict__ Bp,
                                 const float* __restrict__ bias, float* __restrict__ C,
                                 int M, int N) {
    const int lane = threadIdx.x & 31;
    const int wave = threadIdx.x >> 5;
    const int half = lane >> 4;
    const int l16  = lane & 15;
    const int m0   = blockIdx.y * 16;
    const int n0   = blockIdx.x * 256 + wave * 64;
    int row = m0 + l16;
    if (row >= M) row = 0;                 // padded rows: load garbage, never stored

    v8f z = {0.f, 0.f, 0.f, 0.f, 0.f, 0.f, 0.f, 0.f};
    v8f acc[4];
#pragma unroll
    for (int i = 0; i < 4; ++i) acc[i] = z;

    const float* ap0 = A + (size_t)row * KK512 + 8 * half;
    const unsigned int* bp0 = Bp + (size_t)(n0 >> 4) * (KT * 256) + lane * 8;

    for (int kk = 0; kk < KT; ++kk) {
        const unsigned int* bp = bp0 + kk * 256;
        Frag bf[4];
#pragma unroll
        for (int nt = 0; nt < 4; ++nt) {   // all B loads up front (one clause)
            *(uint4*)&bf[nt].u[0] = *(const uint4*)(bp + (size_t)nt * (KT * 256));
            *(uint4*)&bf[nt].u[4] = *(const uint4*)(bp + (size_t)nt * (KT * 256) + 4);
        }

        const float* ap = ap0 + kk * 32;
        float4 a0 = *(const float4*)(ap);
        float4 a1 = *(const float4*)(ap + 4);
        float4 a2 = *(const float4*)(ap + 16);
        float4 a3 = *(const float4*)(ap + 20);
        Frag af;
        af.u[0] = pack_bf16(a0.x, a0.y); af.u[1] = pack_bf16(a0.z, a0.w);
        af.u[2] = pack_bf16(a1.x, a1.y); af.u[3] = pack_bf16(a1.z, a1.w);
        af.u[4] = pack_bf16(a2.x, a2.y); af.u[5] = pack_bf16(a2.z, a2.w);
        af.u[6] = pack_bf16(a3.x, a3.y); af.u[7] = pack_bf16(a3.z, a3.w);

#pragma unroll
        for (int nt = 0; nt < 4; ++nt)
            acc[nt] = WMMA_BF16(af.v, bf[nt].v, acc[nt]);
    }

    const int rbase = m0 + 8 * half;         // C layout: lanes0-15 M=r, lanes16-31 M=8+r
    if (m0 + 16 <= M) {                      // full tile: constant-offset stores
#pragma unroll
        for (int nt = 0; nt < 4; ++nt) {
            int col  = n0 + nt * 16 + l16;
            float bv = bias[col];
            float* cp = C + (size_t)rbase * N + col;
#pragma unroll
            for (int r = 0; r < 8; ++r)
                cp[(size_t)r * N] = acc[nt][r] + bv;
        }
    } else {                                 // ragged last tile
#pragma unroll
        for (int nt = 0; nt < 4; ++nt) {
            int col  = n0 + nt * 16 + l16;
            float bv = bias[col];
            float* cp = C + (size_t)rbase * N + col;
#pragma unroll
            for (int r = 0; r < 8; ++r)
                if (rbase + r < M) cp[(size_t)r * N] = acc[nt][r] + bv;
        }
    }
}

// ---------------------------------------------------------------------------
// LSTM: 1 workgroup, 512 threads (16 waves). Per step:
//   gates(16x2048) = h_bf16(16x512) @ WhhPacked   (rows 8..15 are zero padding)
//   then elementwise i,f,g,o -> c,h ; h -> pred_h (global) + h_bf16 LDS.
// Dynamic LDS: c[8*512] f32 | gates[8*2048] f32 | hb[16*512] bf16 = 98304 B
// ---------------------------------------------------------------------------
__global__ void lstm_kernel(const float* __restrict__ XW,          // (B*U1, 2048) incl. bias
                            const unsigned int* __restrict__ WhhP, // packed (2048,512)
                            float* __restrict__ pred_h) {          // (B*U1, 512)
    extern __shared__ char smem_raw[];
    float*          cs     = (float*)smem_raw;                     // 8*512
    float*          gatesS = cs + 8 * 512;                         // 8*2048
    unsigned short* hb     = (unsigned short*)(gatesS + 8 * 2048); // 16*512

    const int tid  = threadIdx.x;
    const int lane = tid & 31;
    const int wave = tid >> 5;      // 0..15, wave owns gate cols [wave*128, wave*128+128)
    const int half = lane >> 4;
    const int l16  = lane & 15;

    for (int i = tid; i < 8 * 512; i += 512) cs[i] = 0.f;
    for (int i = tid; i < 16 * 512; i += 512) hb[i] = 0;
    __syncthreads();

    v8f z = {0.f, 0.f, 0.f, 0.f, 0.f, 0.f, 0.f, 0.f};
    const unsigned int* bpw = WhhP + (size_t)(wave * 8) * (KT * 256) + lane * 8;

    for (int step = 0; step < U1; ++step) {
        v8f acc[8];
#pragma unroll
        for (int i = 0; i < 8; ++i) acc[i] = z;

        for (int kk = 0; kk < KT; ++kk) {
            const unsigned int* bp = bpw + kk * 256;
            Frag bf[8];
#pragma unroll
            for (int nt = 0; nt < 8; ++nt) {   // all B loads up front
                *(uint4*)&bf[nt].u[0] = *(const uint4*)(bp + (size_t)nt * (KT * 256));
                *(uint4*)&bf[nt].u[4] = *(const uint4*)(bp + (size_t)nt * (KT * 256) + 4);
            }
            Frag af;
#pragma unroll
            for (int r = 0; r < 8; ++r) {
                int k = kk * 32 + ((r < 4) ? 0 : 16) + 2 * (r & 3) + 8 * half;
                af.u[r] = *(const unsigned int*)&hb[l16 * 512 + k];
            }
#pragma unroll
            for (int nt = 0; nt < 8; ++nt)
                acc[nt] = WMMA_BF16(af.v, bf[nt].v, acc[nt]);
        }

        if (half == 0) {   // lanes 0-15 hold batch rows 0-7 (M=r); rows 8-15 are padding
#pragma unroll
            for (int nt = 0; nt < 8; ++nt) {
                int col = wave * 128 + nt * 16 + l16;
#pragma unroll
                for (int r = 0; r < 8; ++r) gatesS[r * 2048 + col] = acc[nt][r];
            }
        }
        __syncthreads();

        // elementwise update: 8 batches * 512 hidden = 4096 elems, 8 per thread
#pragma unroll
        for (int e = 0; e < 8; ++e) {
            int idx = tid + e * 512;
            int b   = idx >> 9;
            int j   = idx & 511;
            const float* xr = XW + ((size_t)b * U1 + step) * 2048;
            float gi = gatesS[b * 2048 + j]        + xr[j];
            float gf = gatesS[b * 2048 + j + 512]  + xr[j + 512];
            float gg = gatesS[b * 2048 + j + 1024] + xr[j + 1024];
            float go = gatesS[b * 2048 + j + 1536] + xr[j + 1536];
            float c  = fsigmoid(gf) * cs[b * 512 + j] + fsigmoid(gi) * ftanh(gg);
            float h  = fsigmoid(go) * ftanh(c);
            cs[b * 512 + j] = c;
            pred_h[((size_t)b * U1 + step) * 512 + j] = h;
            hb[b * 512 + j] = (unsigned short)(pack_bf16(h, 0.f) & 0xFFFFu);
        }
        __syncthreads();
    }
}

// ---------------------------------------------------------------------------
// Fused joint + output GEMM:
//   out[b,t,u,v] = relu(f[b,t,:]+g[b,u,:]) @ out_W^T + out_b    (bf16 WMMA)
// grid = (V/512, 7, B*T), block = 128 (4 waves).
// Wave tile: M=16 (u), N=128 (8 n-tiles). f-row constant per block.
// ---------------------------------------------------------------------------
__global__ void joint_kernel(const float* __restrict__ f,          // (B*T, 512)
                             const float* __restrict__ g,          // (B*U1, 512)
                             const unsigned int* __restrict__ Wp,  // out_W packed
                             const float* __restrict__ ob,         // out_b (1024)
                             float* __restrict__ out) {            // (B,T,U1,V) f32
    const int lane = threadIdx.x & 31;
    const int wave = threadIdx.x >> 5;
    const int half = lane >> 4;
    const int l16  = lane & 15;
    const int bt   = blockIdx.z;                 // 0..1599
    const int b    = bt / TT;
    const int u0   = blockIdx.y * 16;            // 0..96
    const int n0   = blockIdx.x * 512 + wave * 128;

    int urow = u0 + l16;
    if (urow > 100) urow = 100;                  // clamp loads; stores handled below
    const float* fr = f + (size_t)bt * 512 + 8 * half;
    const float* gr = g + ((size_t)b * U1 + urow) * 512 + 8 * half;
    const unsigned int* bp0 = Wp + (size_t)(n0 >> 4) * (KT * 256) + lane * 8;

    v8f z = {0.f, 0.f, 0.f, 0.f, 0.f, 0.f, 0.f, 0.f};
    v8f acc[8];
#pragma unroll
    for (int i = 0; i < 8; ++i) acc[i] = z;

    for (int kk = 0; kk < KT; ++kk) {
        const unsigned int* bp = bp0 + kk * 256;
        Frag bf[8];
#pragma unroll
        for (int nt = 0; nt < 8; ++nt) {       // all 16 B loads up front (one clause)
            *(uint4*)&bf[nt].u[0] = *(const uint4*)(bp + (size_t)nt * (KT * 256));
            *(uint4*)&bf[nt].u[4] = *(const uint4*)(bp + (size_t)nt * (KT * 256) + 4);
        }

        float4 f0 = *(const float4*)(fr + kk * 32);
        float4 f1 = *(const float4*)(fr + kk * 32 + 4);
        float4 f2 = *(const float4*)(fr + kk * 32 + 16);
        float4 f3 = *(const float4*)(fr + kk * 32 + 20);
        float4 g0 = *(const float4*)(gr + kk * 32);
        float4 g1 = *(const float4*)(gr + kk * 32 + 4);
        float4 g2 = *(const float4*)(gr + kk * 32 + 16);
        float4 g3 = *(const float4*)(gr + kk * 32 + 20);
        Frag af;
        af.u[0] = pack_bf16(fmaxf(f0.x + g0.x, 0.f), fmaxf(f0.y + g0.y, 0.f));
        af.u[1] = pack_bf16(fmaxf(f0.z + g0.z, 0.f), fmaxf(f0.w + g0.w, 0.f));
        af.u[2] = pack_bf16(fmaxf(f1.x + g1.x, 0.f), fmaxf(f1.y + g1.y, 0.f));
        af.u[3] = pack_bf16(fmaxf(f1.z + g1.z, 0.f), fmaxf(f1.w + g1.w, 0.f));
        af.u[4] = pack_bf16(fmaxf(f2.x + g2.x, 0.f), fmaxf(f2.y + g2.y, 0.f));
        af.u[5] = pack_bf16(fmaxf(f2.z + g2.z, 0.f), fmaxf(f2.w + g2.w, 0.f));
        af.u[6] = pack_bf16(fmaxf(f3.x + g3.x, 0.f), fmaxf(f3.y + g3.y, 0.f));
        af.u[7] = pack_bf16(fmaxf(f3.z + g3.z, 0.f), fmaxf(f3.w + g3.w, 0.f));

        if (kk + 1 < KT) __builtin_prefetch(bp + 256, 0, 1);   // next k-step B tiles

#pragma unroll
        for (int nt = 0; nt < 8; ++nt)
            acc[nt] = WMMA_BF16(af.v, bf[nt].v, acc[nt]);
    }

    // Epilogue: rows r map to u = u0 + r + 8*half. Blocks with u0 < 96 are full.
    const size_t orow = ((size_t)bt * U1 + u0 + 8 * half) * VV;
    if (u0 < 96) {
#pragma unroll
        for (int nt = 0; nt < 8; ++nt) {
            int col  = n0 + nt * 16 + l16;
            float bv = ob[col];
            float* op = out + orow + col;
#pragma unroll
            for (int r = 0; r < 8; ++r)
                op[(size_t)r * VV] = acc[nt][r] + bv;   // constant-offset stores
        }
    } else {
#pragma unroll
        for (int nt = 0; nt < 8; ++nt) {
            int col  = n0 + nt * 16 + l16;
            float bv = ob[col];
            float* op = out + orow + col;
#pragma unroll
            for (int r = 0; r < 8; ++r)
                if (u0 + 8 * half + r <= 100) op[(size_t)r * VV] = acc[nt][r] + bv;
        }
    }
}

// ---------------------------------------------------------------------------
extern "C" void kernel_launch(void* const* d_in, const int* in_sizes, int n_in,
                              void* d_out, int out_size, void* d_ws, size_t ws_size,
                              hipStream_t stream) {
    const float* enc     = (const float*)d_in[0];   // (8,512,200)
    const int*   targets = (const int*)d_in[2];     // (8,100)
    const float* embed   = (const float*)d_in[4];   // (1024,512)
    const float* Wih     = (const float*)d_in[5];   // (2048,512)
    const float* Whh     = (const float*)d_in[6];   // (2048,512)
    const float* lstm_b  = (const float*)d_in[7];   // (2048)
    const float* enc_W   = (const float*)d_in[8];   // (512,512)
    const float* enc_b   = (const float*)d_in[9];
    const float* pred_W  = (const float*)d_in[10];  // (512,512)
    const float* pred_b  = (const float*)d_in[11];
    const float* out_W   = (const float*)d_in[12];  // (1024,512)
    const float* out_b   = (const float*)d_in[13];
    float* out = (float*)d_out;

    // workspace carve (256B aligned)
    char* ws = (char*)d_ws;
    size_t off = 0;
    auto carve = [&](size_t bytes) { char* p = ws + off; off += (bytes + 255) & ~(size_t)255; return p; };
    unsigned int* pk_outW  = (unsigned int*)carve((size_t)VV  / 16 * KT * 256 * 4); // 1 MB
    unsigned int* pk_whh   = (unsigned int*)carve((size_t)G4H / 16 * KT * 256 * 4); // 2 MB
    unsigned int* pk_wih   = (unsigned int*)carve((size_t)G4H / 16 * KT * 256 * 4); // 2 MB
    unsigned int* pk_encW  = (unsigned int*)carve((size_t)HH  / 16 * KT * 256 * 4);
    unsigned int* pk_predW = (unsigned int*)carve((size_t)HH  / 16 * KT * 256 * 4);
    float* encT   = (float*)carve((size_t)BB * TT * HH * 4);   // (1600,512)
    float* x      = (float*)carve((size_t)BB * U1 * HH * 4);   // (808,512)
    float* XW     = (float*)carve((size_t)BB * U1 * G4H * 4);  // (808,2048)
    float* pred_h = (float*)carve((size_t)BB * U1 * HH * 4);   // (808,512)
    float* ff     = (float*)carve((size_t)BB * TT * HH * 4);   // (1600,512)
    float* gg     = (float*)carve((size_t)BB * U1 * HH * 4);   // (808,512)
    (void)ws_size; (void)in_sizes; (void)n_in; (void)out_size;

    // 1) pack weights into WMMA B-fragment layout (L2-resident thereafter)
    pack_b_kernel<<<dim3(VV / 16,  KT), 32, 0, stream>>>(out_W,  pk_outW,  VV,  KK512);
    pack_b_kernel<<<dim3(G4H / 16, KT), 32, 0, stream>>>(Whh,    pk_whh,   G4H, KK512);
    pack_b_kernel<<<dim3(G4H / 16, KT), 32, 0, stream>>>(Wih,    pk_wih,   G4H, KK512);
    pack_b_kernel<<<dim3(HH / 16,  KT), 32, 0, stream>>>(enc_W,  pk_encW,  HH,  KK512);
    pack_b_kernel<<<dim3(HH / 16,  KT), 32, 0, stream>>>(pred_W, pk_predW, HH,  KK512);

    // 2) embedding gather (+SOS) and encoder transpose
    {
        long n = (long)BB * U1 * HH;
        build_x_kernel<<<(unsigned)((n + 255) / 256), 256, 0, stream>>>(targets, embed, x);
    }
    {
        long n = (long)BB * TT * HH;
        transpose_enc_kernel<<<(unsigned)((n + 255) / 256), 256, 0, stream>>>(enc, encT);
    }

    // 3) XW = x @ Wih^T + lstm_b   (M=808, N=2048)
    gemm_bf16_kernel<<<dim3(G4H / 256, (BB * U1 + 15) / 16), 128, 0, stream>>>(
        x, pk_wih, lstm_b, XW, BB * U1, G4H);

    // 4) f = encT @ enc_W^T + enc_b   (M=1600, N=512)
    gemm_bf16_kernel<<<dim3(HH / 256, (BB * TT) / 16), 128, 0, stream>>>(
        encT, pk_encW, enc_b, ff, BB * TT, HH);

    // 5) LSTM (sequential, single WGP, WMMA recurrence)
    lstm_kernel<<<1, 512, 8 * 512 * 4 + 8 * 2048 * 4 + 16 * 512 * 2, stream>>>(XW, pk_whh, pred_h);

    // 6) g = pred_h @ pred_W^T + pred_b   (M=808, N=512)
    gemm_bf16_kernel<<<dim3(HH / 256, (BB * U1 + 15) / 16), 128, 0, stream>>>(
        pred_h, pk_predW, pred_b, gg, BB * U1, HH);

    // 7) fused joint + output projection (bandwidth-bound: 662 MB of logits)
    joint_kernel<<<dim3(VV / 512, 7, BB * TT), 128, 0, stream>>>(ff, gg, pk_outW, out_b, out);
}
